// Head_23398981828932
// MI455X (gfx1250) — matrix operand order repaired and checked by reading, not compiled, not run on hardware
//
#include <hip/hip_runtime.h>

#define SEQ   100
#define NE    1001
#define NEP   1024   // K padded to multiple of 32
#define HEAD  64
#define SEQP  112    // 7*16
#define SCOL  128    // score/prob cols padded to multiple of 32

typedef __attribute__((ext_vector_type(16))) __bf16 v16bf;
typedef __attribute__((ext_vector_type(8)))  __bf16 v8bf;
typedef __attribute__((ext_vector_type(8)))  float  v8f;

__device__ __forceinline__ v8f wmma_bf16(v16bf a, v16bf b, v8f c) {
  // D = A(16x32 bf16) * B(32x16 bf16) + C(16x16 f32)
  return __builtin_amdgcn_wmma_f32_16x16x32_bf16(false, a, false, b, (short)0, c,
                                                 false, false);
}

__device__ __forceinline__ v16bf combine8(v8bf lo, v8bf hi) {
  v16bf r;
#pragma unroll
  for (int i = 0; i < 8; ++i) { r[i] = lo[i]; r[i + 8] = hi[i]; }
  return r;
}

// A-fragment (16x32, 16-bit): lane half (lane>>4) picks K runs
//   elems 0..7  -> K = kBase + half*8 .. +7
//   elems 8..15 -> K = kBase + 16 + half*8 .. +7
__device__ __forceinline__ v16bf a_frag_global(const float* __restrict__ row,
                                               int kBase, int half, bool tail) {
  v16bf a;
  const int k0 = kBase + half * 8;
  const int k1 = k0 + 16;
  if (!tail) {
#pragma unroll
    for (int i = 0; i < 8; ++i) {
      a[i]     = (__bf16)row[k0 + i];
      a[i + 8] = (__bf16)row[k1 + i];
    }
  } else {
#pragma unroll
    for (int i = 0; i < 8; ++i) {
      a[i]     = (__bf16)((k0 + i < NE) ? row[k0 + i] : 0.0f);
      a[i + 8] = (__bf16)((k1 + i < NE) ? row[k1 + i] : 0.0f);
    }
  }
  return a;
}

// A-fragment from bf16 LDS row (16B-aligned runs)
__device__ __forceinline__ v16bf a_frag_lds(const __bf16* row, int kBase, int half) {
  const int k0 = kBase + half * 8;
  return combine8(*(const v8bf*)(row + k0), *(const v8bf*)(row + k0 + 16));
}

// B-fragment (32x16, 16-bit): lane n = lane&15; p = base + half*16;
// elems 0..7 -> K = half*8 .. +7, elems 8..15 -> K = 16 + half*8 .. +7.
__device__ __forceinline__ v16bf b_frag(const __bf16* p) {
  const __bf16* r = p - ((threadIdx.x & 31) >> 4) * 8;
  return combine8(*(const v8bf*)r, *(const v8bf*)(r + 16));
}

// ---------------- Kernel 0: weights f32[HEAD][NE] -> bf16[3][HEAD][NEP] (zero pad)
__global__ void wconvert_kernel(const float* __restrict__ Wk,
                                const float* __restrict__ Wq,
                                const float* __restrict__ Wv,
                                __bf16* __restrict__ outw) {
  const int idx = blockIdx.x * blockDim.x + threadIdx.x;
  if (idx >= 3 * HEAD * NEP) return;
  const int m = idx / (HEAD * NEP);
  const int r = (idx / NEP) % HEAD;
  const int k = idx % NEP;
  const float* W = (m == 0) ? Wk : ((m == 1) ? Wq : Wv);
  const float v = (k < NE) ? W[(size_t)r * NE + k] : 0.0f;
  outw[idx] = (__bf16)v;
}

// ---------------- Kernel 1: fused attention head, one block per batch element
__global__ __launch_bounds__(256) void head_attn_kernel(
    const float* __restrict__ x1, const float* __restrict__ x2,
    const __bf16* __restrict__ wb /* [3][HEAD][NEP] bf16: Wk,Wq,Wv */,
    float* __restrict__ out /* [B][SEQ][HEAD] f32 */) {
  __shared__ __align__(32) __bf16 Kl[SEQP][HEAD];   // keys, row-major
  __shared__ __align__(32) __bf16 Ql[SEQP][HEAD];   // queries, row-major
  __shared__ __align__(32) __bf16 Vt[HEAD][SCOL];   // values, TRANSPOSED [h][seq]
  __shared__ __align__(32) float  Sl[SEQP][SCOL];   // scores f32
  __shared__ __align__(32) __bf16 Pl[SEQP][SCOL];   // softmax probs bf16

  const int b     = blockIdx.x;
  const int tid   = threadIdx.x;
  // wave index is uniform across the wave -- force it scalar so all job loops
  // and the kJob branch are SALU control flow (EXEC provably all-ones at WMMA).
  const int wave  = __builtin_amdgcn_readfirstlane(tid >> 5);
  const int lane  = tid & 31;
  const int lhalf = lane >> 4;   // 0|1
  const int l16   = lane & 15;

  // ======= Phase 1: K = x1*Wk^T ; {Q,V} = x2*{Wq,Wv}^T (x2 read ONCE) =======
  // jobs 0..6  : K projection, strip = job       (x1, Wk, 4 accumulators)
  // jobs 7..13 : fused Q+V,    strip = job - 7   (x2, Wq+Wv, 8 accumulators)
  for (int job = wave; job < 14; job += 8) {
    const bool kJob  = (job < 7);
    const int  strip = kJob ? job : (job - 7);
    const float* xs  = kJob ? x1 : x2;

    const int mRow = strip * 16 + l16;
    const int mCl  = (mRow < SEQ) ? mRow : (SEQ - 1);
    const float* arow = xs + ((size_t)b * SEQ + mCl) * NE;

    const __bf16* w0 = wb + (kJob ? 0 : (size_t)1 * HEAD * NEP); // Wk or Wq
    const __bf16* w1 = wb + (size_t)2 * HEAD * NEP;              // Wv (QV job only)

    v8f q0 = {0,0,0,0,0,0,0,0}, q1 = {0,0,0,0,0,0,0,0};
    v8f q2 = {0,0,0,0,0,0,0,0}, q3 = {0,0,0,0,0,0,0,0};
    v8f v0 = {0,0,0,0,0,0,0,0}, v1 = {0,0,0,0,0,0,0,0};
    v8f v2 = {0,0,0,0,0,0,0,0}, v3 = {0,0,0,0,0,0,0,0};

    for (int kB = 0; kB < NEP; kB += 32) {
      const bool tail = (kB + 32 > NE);
      v16bf af = a_frag_global(arow, kB, lhalf, tail);
      const int koff = kB + lhalf * 16;
      {
        v16bf b0 = b_frag(w0 + (size_t)(l16 +  0) * NEP + koff);
        v16bf b1 = b_frag(w0 + (size_t)(l16 + 16) * NEP + koff);
        v16bf b2 = b_frag(w0 + (size_t)(l16 + 32) * NEP + koff);
        v16bf b3 = b_frag(w0 + (size_t)(l16 + 48) * NEP + koff);
        q0 = wmma_bf16(af, b0, q0);
        q1 = wmma_bf16(af, b1, q1);
        q2 = wmma_bf16(af, b2, q2);
        q3 = wmma_bf16(af, b3, q3);
      }
      if (!kJob) {   // scalar branch: kJob depends only on uniform `wave`
        v16bf b0 = b_frag(w1 + (size_t)(l16 +  0) * NEP + koff);
        v16bf b1 = b_frag(w1 + (size_t)(l16 + 16) * NEP + koff);
        v16bf b2 = b_frag(w1 + (size_t)(l16 + 32) * NEP + koff);
        v16bf b3 = b_frag(w1 + (size_t)(l16 + 48) * NEP + koff);
        v0 = wmma_bf16(af, b0, v0);
        v1 = wmma_bf16(af, b1, v1);
        v2 = wmma_bf16(af, b2, v2);
        v3 = wmma_bf16(af, b3, v3);
      }
    }
#pragma unroll
    for (int v = 0; v < 8; ++v) {
      const int m = strip * 16 + v + lhalf * 8;  // C-frag row
      if (m < SEQ) {
        if (kJob) {
          Kl[m][ 0 + l16] = (__bf16)q0[v];
          Kl[m][16 + l16] = (__bf16)q1[v];
          Kl[m][32 + l16] = (__bf16)q2[v];
          Kl[m][48 + l16] = (__bf16)q3[v];
        } else {
          Ql[m][ 0 + l16] = (__bf16)q0[v];
          Ql[m][16 + l16] = (__bf16)q1[v];
          Ql[m][32 + l16] = (__bf16)q2[v];
          Ql[m][48 + l16] = (__bf16)q3[v];
          Vt[ 0 + l16][m] = (__bf16)v0[v];
          Vt[16 + l16][m] = (__bf16)v1[v];
          Vt[32 + l16][m] = (__bf16)v2[v];
          Vt[48 + l16][m] = (__bf16)v3[v];
        }
      }
    }
  }
  __syncthreads();

  // ======= Phase 2: S = (Q * K^T) * 0.125  (K-dim = HEAD = 2 chunks) =======
  for (int job = wave; job < 49; job += 8) {
    const int mt = job / 7, nt = job % 7;
    v8f acc = {0,0,0,0,0,0,0,0};
#pragma unroll
    for (int h = 0; h < HEAD; h += 32) {
      v16bf af = a_frag_lds(&Ql[mt * 16 + l16][0], h, lhalf);
      v16bf bf = b_frag(&Kl[nt * 16 + l16][h + lhalf * 16]);
      acc = wmma_bf16(af, bf, acc);
    }
#pragma unroll
    for (int v = 0; v < 8; ++v)
      Sl[mt * 16 + v + lhalf * 8][nt * 16 + l16] = acc[v] * 0.125f;
  }
  __syncthreads();

  // ======= Phase 3: row softmax (mask==ones -> only pad cols excluded) =======
  if (tid < SEQP) {
    if (tid < SEQ) {
      const float* srow = &Sl[tid][0];
      float mx = -3.402823466e38f;
      for (int j = 0; j < SEQ; ++j) mx = fmaxf(mx, srow[j]);
      float sum = 0.0f;
      for (int j = 0; j < SEQ; ++j) sum += __expf(srow[j] - mx);
      const float inv = 1.0f / sum;
      for (int j = 0; j < SEQ; ++j) Pl[tid][j] = (__bf16)(__expf(srow[j] - mx) * inv);
      for (int j = SEQ; j < SCOL; ++j) Pl[tid][j] = (__bf16)0.0f;
    } else {
      for (int j = 0; j < SCOL; ++j) Pl[tid][j] = (__bf16)0.0f;
    }
  }
  __syncthreads();

  // ======= Phase 4: out = P * V   (K-dim = SCOL = 4 chunks; V transposed) =======
  for (int job = wave; job < 28; job += 8) {
    const int strip = job / 4, nt = job % 4;
    const __bf16* prow = &Pl[strip * 16 + l16][0];
    v8f acc = {0,0,0,0,0,0,0,0};
#pragma unroll
    for (int jB = 0; jB < SCOL; jB += 32) {
      v16bf af = a_frag_lds(prow, jB, lhalf);
      v16bf bf = b_frag(&Vt[nt * 16 + l16][jB + lhalf * 16]);
      acc = wmma_bf16(af, bf, acc);
    }
#pragma unroll
    for (int v = 0; v < 8; ++v) {
      const int i = strip * 16 + v + lhalf * 8;
      if (i < SEQ)
        out[((size_t)b * SEQ + i) * HEAD + nt * 16 + l16] = acc[v];
    }
  }
}

extern "C" void kernel_launch(void* const* d_in, const int* in_sizes, int n_in,
                              void* d_out, int out_size, void* d_ws, size_t ws_size,
                              hipStream_t stream) {
  const float* x1 = (const float*)d_in[0];
  const float* x2 = (const float*)d_in[1];
  const float* Wk = (const float*)d_in[2];
  const float* Wq = (const float*)d_in[3];
  const float* Wv = (const float*)d_in[4];
  float* out = (float*)d_out;

  // workspace: 3*HEAD*NEP bf16 = 384 KB of padded bf16 weights
  __bf16* wb = (__bf16*)d_ws;

  const int wtot = 3 * HEAD * NEP;
  wconvert_kernel<<<(wtot + 255) / 256, 256, 0, stream>>>(Wk, Wq, Wv, wb);
  head_attn_kernel<<<1024, 256, 0, stream>>>(x1, x2, wb, out);
}
